// GATv2WithGlobal_4389456577283
// MI455X (gfx1250) — matrix operand order repaired and checked by reading, not compiled, not run on hardware
//
#include <hip/hip_runtime.h>
#include <hip/hip_bf16.h>

#define NNODES 20000
#define NEDGES 320000
#define ETOT   (NEDGES + NNODES)   // 340000 (edges + self loops)
#define NGRAPH 64
#define NEG_SLOPE 0.2f
#define BN_EPS 1e-5f

typedef __attribute__((ext_vector_type(16))) __bf16 v16bf;
typedef __attribute__((ext_vector_type(8)))  float  v8f;

// ---------- helpers ----------
__device__ __forceinline__ unsigned short f2bf(float f) {  // RNE f32 -> bf16
    unsigned u = __float_as_uint(f);
    u += 0x7FFFu + ((u >> 16) & 1u);
    return (unsigned short)(u >> 16);
}
// order-preserving float <-> uint map (for atomicMax on floats incl. negatives)
__device__ __forceinline__ unsigned ordF(float f) {
    unsigned b = __float_as_uint(f);
    return (b & 0x80000000u) ? ~b : (b | 0x80000000u);
}
__device__ __forceinline__ float unordF(unsigned u) {
    return __uint_as_float((u & 0x80000000u) ? (u & 0x7FFFFFFFu) : ~u);
}

// ---------- layer 1 linear: x[20000,9] @ W[9,256] + b (both l and r) ----------
__global__ void k_lin1(const float* __restrict__ x,
                       const float* __restrict__ Wl, const float* __restrict__ bl,
                       const float* __restrict__ Wr, const float* __restrict__ br,
                       float* __restrict__ xl, float* __restrict__ xr) {
    int n = blockIdx.x;          // node
    int c = threadIdx.x;         // 256 channels
    __shared__ float xs[9];
    if (c < 9) xs[c] = x[n * 9 + c];
    __syncthreads();
    float sl = bl[c], sr = br[c];
#pragma unroll
    for (int k = 0; k < 9; ++k) {
        float xv = xs[k];
        sl += xv * Wl[k * 256 + c];
        sr += xv * Wr[k * 256 + c];
    }
    xl[(size_t)n * 256 + c] = sl;
    xr[(size_t)n * 256 + c] = sr;
}

// ---------- edge logits: e[edge][h] = sum_c lrelu(xl[src]+xr[dst]) * att ----------
template <int H, int C, int HC>
__global__ void k_edge_e(const int* __restrict__ ei,
                         const float* __restrict__ xl, const float* __restrict__ xr,
                         const float* __restrict__ att, float* __restrict__ e) {
    int wib  = threadIdx.x >> 5;
    int lane = threadIdx.x & 31;
    long long eid = (long long)blockIdx.x * (blockDim.x >> 5) + wib;  // wave per edge
    if (eid >= ETOT) return;
    int src, dst;
    if (eid < NEDGES) { src = ei[eid]; dst = ei[NEDGES + eid]; }
    else              { src = dst = (int)(eid - NEDGES); }
    const float* pl = xl + (size_t)src * HC;
    const float* pr = xr + (size_t)dst * HC;
#pragma unroll
    for (int h = 0; h < H; ++h) {
        float s = 0.f;
#pragma unroll
        for (int i = 0; i < C / 32; ++i) {
            int c = h * C + i * 32 + lane;
            float m = pl[c] + pr[c];
            m = (m > 0.f) ? m : NEG_SLOPE * m;
            s += m * att[c];
        }
        for (int off = 16; off > 0; off >>= 1) s += __shfl_xor(s, off, 32);
        if (lane == 0) e[eid * H + h] = s;
    }
}

// ---------- segment max over dst (ordered-uint atomicMax) ----------
template <int H>
__global__ void k_emax(const int* __restrict__ ei, const float* __restrict__ e,
                       unsigned* __restrict__ emax) {
    long long t = (long long)blockIdx.x * blockDim.x + threadIdx.x;
    if (t >= (long long)ETOT * H) return;
    int eid = (int)(t / H);
    int dst = (eid < NEDGES) ? ei[NEDGES + eid] : eid - NEDGES;
    atomicMax(&emax[dst * H + (int)(t % H)], ordF(e[t]));
}

// ---------- ex = exp(e - emax[dst]); denom[dst] += ex ----------
template <int H>
__global__ void k_expsum(const int* __restrict__ ei, float* __restrict__ e,
                         const unsigned* __restrict__ emax, float* __restrict__ denom) {
    long long t = (long long)blockIdx.x * blockDim.x + threadIdx.x;
    if (t >= (long long)ETOT * H) return;
    int eid = (int)(t / H);
    int h   = (int)(t % H);
    int dst = (eid < NEDGES) ? ei[NEDGES + eid] : eid - NEDGES;
    float ex = __expf(e[t] - unordF(emax[dst * H + h]));
    e[t] = ex;
    atomicAdd(&denom[dst * H + h], ex);
}

// ---------- aggregate: agg[dst] += alpha * xl[src] ----------
template <int H, int C, int HC>
__global__ void k_agg(const int* __restrict__ ei, const float* __restrict__ ex,
                      const float* __restrict__ denom, const float* __restrict__ xl,
                      float* __restrict__ agg) {
    int eid = blockIdx.x;
    int src, dst;
    if (eid < NEDGES) { src = ei[eid]; dst = ei[NEDGES + eid]; }
    else              { src = dst = eid - NEDGES; }
    for (int cc = threadIdx.x; cc < HC; cc += blockDim.x) {
        int h = cc / C;
        float alpha = ex[(size_t)eid * H + h] / denom[dst * H + h];
        atomicAdd(&agg[(size_t)dst * HC + cc], alpha * xl[(size_t)src * HC + cc]);
    }
}

// ---------- BN + ReLU + bf16 convert (layer1 output -> GEMM A matrix) ----------
__global__ void k_bn1(const float* __restrict__ agg, const float* __restrict__ bias,
                      const float* __restrict__ g, const float* __restrict__ b,
                      const float* __restrict__ mean, const float* __restrict__ var,
                      unsigned short* __restrict__ h1b) {
    size_t i = (size_t)blockIdx.x * blockDim.x + threadIdx.x;
    if (i >= (size_t)NNODES * 256) return;
    int c = (int)(i & 255);
    float v = agg[i] + bias[c];
    v = (v - mean[c]) * rsqrtf(var[c] + BN_EPS) * g[c] + b[c];
    h1b[i] = f2bf(fmaxf(v, 0.f));
}

// ---------- pack W[256,512] fp32 -> bf16 in exact WMMA B-fragment order ----------
// P[kt<<14 | nt<<9 | lane<<4 | j] = W[kt*32 + (lane>=16?8:0) + (j&7) + (j>=8?16:0)][nt*16 + (lane&15)]
__global__ void k_pack(const float* __restrict__ W, unsigned short* __restrict__ P) {
    int idx = blockIdx.x * blockDim.x + threadIdx.x;  // < 256*512
    if (idx >= 256 * 512) return;
    int j    = idx & 15;
    int lane = (idx >> 4) & 31;
    int nt   = (idx >> 9) & 31;
    int kt   = idx >> 14;
    int k = kt * 32 + ((lane & 16) ? 8 : 0) + (j & 7) + ((j & 8) ? 16 : 0);
    int n = nt * 16 + (lane & 15);
    P[idx] = f2bf(W[k * 512 + n]);
}

// ---------- layer-2 GEMM via WMMA: [20000,256]bf16 @ [256,512] -> xl2 & xr2 ----------
// One wave per 16x16 output tile; K looped 8x32 with v_wmma_f32_16x16x32_bf16.
__global__ __launch_bounds__(256) void k_gemm2(
        const unsigned short* __restrict__ A,
        const unsigned short* __restrict__ Pl, const float* __restrict__ bl,
        const unsigned short* __restrict__ Pr, const float* __restrict__ br,
        float* __restrict__ xl2, float* __restrict__ xr2) {
    int wave = (blockIdx.x * blockDim.x + threadIdx.x) >> 5;
    int lane = threadIdx.x & 31;
    int nt   = wave & 31;          // 32 N tiles
    int side = (wave >> 5) & 1;    // 0 = xl (W_l2), 1 = xr (W_r2)
    int mt   = wave >> 6;          // 1250 M tiles
    if (mt >= 1250) return;
    const unsigned short* P    = side ? Pr : Pl;
    const float*          bias = side ? br : bl;
    float*                out  = side ? xr2 : xl2;

    int row   = mt * 16 + (lane & 15);
    int kbase = (lane & 16) ? 8 : 0;                  // A 16x32 bf16 fragment pattern
    const unsigned short* Arow = A + (size_t)row * 256;

    v8f acc = {0.f, 0.f, 0.f, 0.f, 0.f, 0.f, 0.f, 0.f};
#pragma unroll
    for (int kt = 0; kt < 8; ++kt) {
        int k0 = kt * 32;
        union { v16bf v; uint4 q[2]; } af, bfr;
        af.q[0] = *(const uint4*)(Arow + k0 + kbase);        // K = k0+kbase .. +7
        af.q[1] = *(const uint4*)(Arow + k0 + kbase + 16);   // K = k0+kbase+16 .. +23
        const unsigned short* Bp = P + (((size_t)kt << 14) + (nt << 9) + (lane << 4));
        bfr.q[0] = *(const uint4*)(Bp);
        bfr.q[1] = *(const uint4*)(Bp + 8);
        acc = __builtin_amdgcn_wmma_f32_16x16x32_bf16(
            /*neg_a=*/false, af.v, /*neg_b=*/false, bfr.v,
            /*c_mod=*/(short)0, acc, /*reuse_a=*/false, /*reuse_b=*/false);
    }
    int n  = nt * 16 + (lane & 15);
    float bv = bias[n];
    int m0 = mt * 16 + ((lane & 16) ? 8 : 0);         // C/D layout: VGPR r -> M = m0+r
#pragma unroll
    for (int r = 0; r < 8; ++r)
        out[(size_t)(m0 + r) * 512 + n] = acc[r] + bv;
}

// ---------- BN + ReLU in place (layer2) ----------
__global__ void k_bn2(float* __restrict__ agg, const float* __restrict__ bias,
                      const float* __restrict__ g, const float* __restrict__ b,
                      const float* __restrict__ mean, const float* __restrict__ var) {
    size_t i = (size_t)blockIdx.x * blockDim.x + threadIdx.x;
    if (i >= (size_t)NNODES * 512) return;
    int c = (int)(i & 511);
    float v = agg[i] + bias[c];
    v = (v - mean[c]) * rsqrtf(var[c] + BN_EPS) * g[c] + b[c];
    agg[i] = fmaxf(v, 0.f);
}

// ---------- global mean pool ----------
__global__ void k_pool(const float* __restrict__ h2, const int* __restrict__ batch,
                       float* __restrict__ pooled, float* __restrict__ cnt) {
    size_t i = (size_t)blockIdx.x * blockDim.x + threadIdx.x;
    if (i >= (size_t)NNODES * 512) return;
    int n = (int)(i >> 9);
    int c = (int)(i & 511);
    int gidx = batch[n];
    atomicAdd(&pooled[gidx * 512 + c], h2[i]);
    if (c == 0) atomicAdd(&cnt[gidx], 1.f);
}

__global__ void k_z(const float* __restrict__ pooled, const float* __restrict__ cnt,
                    const float* __restrict__ gf, float* __restrict__ z) {
    int i = blockIdx.x * blockDim.x + threadIdx.x;
    if (i >= NGRAPH * 532) return;
    int g = i / 532, c = i % 532;
    z[i] = (c < 512) ? pooled[g * 512 + c] / fmaxf(cnt[g], 1.f)
                     : gf[g * 20 + (c - 512)];
}

__global__ void k_fc1(const float* __restrict__ z, const float* __restrict__ W,
                      const float* __restrict__ b, float* __restrict__ o) {
    int i = blockIdx.x * blockDim.x + threadIdx.x;
    if (i >= NGRAPH * 128) return;
    int g = i >> 7, c = i & 127;
    float s = b[c];
    for (int k = 0; k < 532; ++k) s += z[g * 532 + k] * W[k * 128 + c];
    o[i] = fmaxf(s, 0.f);
}

__global__ void k_fc2(const float* __restrict__ o, const float* __restrict__ W,
                      const float* __restrict__ b, float* __restrict__ out) {
    int g = blockIdx.x * blockDim.x + threadIdx.x;
    if (g >= NGRAPH) return;
    float s = b[0];
    for (int k = 0; k < 128; ++k) s += o[g * 128 + k] * W[k];
    out[g] = s;
}

// =======================================================================
extern "C" void kernel_launch(void* const* d_in, const int* in_sizes, int n_in,
                              void* d_out, int out_size, void* d_ws, size_t ws_size,
                              hipStream_t stream) {
    (void)in_sizes; (void)n_in; (void)out_size; (void)ws_size;
    const float* x    = (const float*)d_in[0];
    const float* gf   = (const float*)d_in[1];
    const int*   ei   = (const int*)d_in[2];
    const int*   batch= (const int*)d_in[3];
    const float* Wl1  = (const float*)d_in[4];
    const float* bl1  = (const float*)d_in[5];
    const float* Wr1  = (const float*)d_in[6];
    const float* br1  = (const float*)d_in[7];
    const float* att1 = (const float*)d_in[8];
    const float* bias1= (const float*)d_in[9];
    const float* g1   = (const float*)d_in[10];
    const float* be1  = (const float*)d_in[11];
    const float* mu1  = (const float*)d_in[12];
    const float* va1  = (const float*)d_in[13];
    const float* Wl2  = (const float*)d_in[14];
    const float* bl2  = (const float*)d_in[15];
    const float* Wr2  = (const float*)d_in[16];
    const float* br2  = (const float*)d_in[17];
    const float* att2 = (const float*)d_in[18];
    const float* bias2= (const float*)d_in[19];
    const float* g2   = (const float*)d_in[20];
    const float* be2  = (const float*)d_in[21];
    const float* mu2  = (const float*)d_in[22];
    const float* va2  = (const float*)d_in[23];
    const float* fc1W = (const float*)d_in[24];
    const float* fc1b = (const float*)d_in[25];
    const float* fc2W = (const float*)d_in[26];
    const float* fc2b = (const float*)d_in[27];
    float* out = (float*)d_out;

    char* ws = (char*)d_ws;
    const size_t MB = 1ull << 20;
    // region layout (with cross-layer reuse):
    float*          xl1   = (float*)(ws + 0);            // 20.48 MB  [region0]
    float*          xr1   = (float*)(ws + 21 * MB);      // 20.48 MB  [region0]
    float*          xl2   = (float*)(ws + 0);            // 40.96 MB  reuses region0
    float*          agg1  = (float*)(ws + 44 * MB);      // 20.48 MB  [region1]
    float*          xr2   = (float*)(ws + 44 * MB);      // 40.96 MB  reuses region1
    unsigned short* h1b   = (unsigned short*)(ws + 86 * MB);   // 10.24 MB
    float*          agg2  = (float*)(ws + 97 * MB);      // 40.96 MB (BN'd in place -> h2)
    float*          ebuf  = (float*)(ws + 137 * MB);     // ETOT*4 floats = 5.44 MB
    unsigned*       emax  = (unsigned*)(ws + 143 * MB);  // 320 KB
    float*          denom = (float*)(ws + 143 * MB + 512 * 1024); // 320 KB
    unsigned short* Plk   = (unsigned short*)(ws + 144 * MB);           // 256 KB
    unsigned short* Prk   = (unsigned short*)(ws + 144 * MB + 512 * 1024); // 256 KB
    float*          pooled= (float*)(ws + 145 * MB);     // 128 KB
    float*          cnt   = (float*)(ws + 145 * MB + 256 * 1024); // 256 B
    float*          zbuf  = (float*)(ws + 145 * MB + 512 * 1024); // 136 KB
    float*          fc1o  = (float*)(ws + 146 * MB);     // 32 KB

    const long long EH = (long long)ETOT * 4;

    // ---- layer 1 ----
    hipMemsetAsync(agg1, 0, (size_t)NNODES * 256 * 4, stream);
    hipMemsetAsync(emax, 0, (size_t)NNODES * 4 * 4, stream);   // ordF(-inf) floor is 0-safe
    hipMemsetAsync(denom, 0, (size_t)NNODES * 4 * 4, stream);
    k_lin1<<<NNODES, 256, 0, stream>>>(x, Wl1, bl1, Wr1, br1, xl1, xr1);
    k_edge_e<4, 64, 256><<<(ETOT + 7) / 8, 256, 0, stream>>>(ei, xl1, xr1, att1, ebuf);
    k_emax<4><<<(int)((EH + 255) / 256), 256, 0, stream>>>(ei, ebuf, emax);
    k_expsum<4><<<(int)((EH + 255) / 256), 256, 0, stream>>>(ei, ebuf, emax, denom);
    k_agg<4, 64, 256><<<ETOT, 256, 0, stream>>>(ei, ebuf, denom, xl1, agg1);
    k_bn1<<<NNODES, 256, 0, stream>>>(agg1, bias1, g1, be1, mu1, va1, h1b);

    // ---- layer 2 dense transforms (WMMA) ----
    k_pack<<<512, 256, 0, stream>>>(Wl2, Plk);
    k_pack<<<512, 256, 0, stream>>>(Wr2, Prk);
    // waves = 1250 Mtiles * 32 Ntiles * 2 sides = 80000 -> 10000 blocks of 8 waves
    k_gemm2<<<10000, 256, 0, stream>>>(h1b, Plk, bl2, Prk, br2, xl2, xr2);

    // ---- layer 2 attention ----
    hipMemsetAsync(agg2, 0, (size_t)NNODES * 512 * 4, stream);
    hipMemsetAsync(emax, 0, (size_t)NNODES * 4 * 4, stream);
    hipMemsetAsync(denom, 0, (size_t)NNODES * 4 * 4, stream);
    k_edge_e<4, 128, 512><<<(ETOT + 7) / 8, 256, 0, stream>>>(ei, xl2, xr2, att2, ebuf);
    k_emax<4><<<(int)((EH + 255) / 256), 256, 0, stream>>>(ei, ebuf, emax);
    k_expsum<4><<<(int)((EH + 255) / 256), 256, 0, stream>>>(ei, ebuf, emax, denom);
    k_agg<4, 128, 512><<<ETOT, 256, 0, stream>>>(ei, ebuf, denom, xl2, agg2);
    k_bn2<<<NNODES * 2, 256, 0, stream>>>(agg2, bias2, g2, be2, mu2, va2);

    // ---- pool + head ----
    hipMemsetAsync(pooled, 0, (size_t)NGRAPH * 512 * 4, stream);
    hipMemsetAsync(cnt, 0, (size_t)NGRAPH * 4, stream);
    k_pool<<<NNODES * 2, 256, 0, stream>>>(agg2, batch, pooled, cnt);
    k_z<<<(NGRAPH * 532 + 255) / 256, 256, 0, stream>>>(pooled, cnt, gf, zbuf);
    k_fc1<<<(NGRAPH * 128 + 255) / 256, 256, 0, stream>>>(zbuf, fc1W, fc1b, fc1o);
    k_fc2<<<1, 64, 0, stream>>>(fc1o, fc2W, fc2b, out);
}
